// BayesianFilter_46772193854007
// MI455X (gfx1250) — matrix operand order repaired and checked by reading, not compiled, not run on hardware
//
#include <hip/hip_runtime.h>
#include <hip/hip_bf16.h>

typedef __attribute__((ext_vector_type(16))) __bf16 v16bf;
typedef __attribute__((ext_vector_type(8)))  float  v8f;

#define N_   8
#define V_   2048
#define NV_  16384
#define C_   16
#define T_   150
#define L_   64
#define E_   32768

#define WAVES_ 4
#define BLK_   128
#define NPB_   64   // nodes per block (16 per wave)

// bf16 transposed-weight pool offsets (elements)
#define OFF_ENC    0        // 64 rows x ld 32 (K=16 padded to 32)
#define OFF_GD     2048     // 64 x 64
#define OFF_GDWI   6144     // 192 x 64
#define OFF_GDWH   18432    // 192 x 64
#define OFF_DOM1   30720    // 64 x 64
#define OFF_DOM2   34816    // 64 x 64
#define OFF_ODE1   38912    // 64 x 128
#define OFF_ODE2   47104    // 64 x 64
#define OFF_GC     51200    // 64 x 64
#define OFF_GCWI   55296    // 192 x 64
#define OFF_GCWH   67584    // 192 x 64
#define OFF_DEC    79872    // 16 x 64
#define BW_ELEMS   80896

#define WMMA(A,B,Cv) __builtin_amdgcn_wmma_f32_16x16x32_bf16(false,(A),false,(B),(short)0,(Cv),false,false)

__device__ inline v8f vzero() {
  v8f z;
#pragma unroll
  for (int i = 0; i < 8; ++i) z[i] = 0.0f;
  return z;
}

__device__ inline float sigm(float v) { return 1.0f / (1.0f + __expf(-v)); }

// prefetch the (L2-resident) bf16 weight pool: lowers to global_prefetch_b8
__device__ inline void prefetch_weights(const __bf16* bw) {
  // two cacheline-granular prefetches per thread (at +0 and +160 B), striding the ~160KB pool
  const char* p = (const char*)bw + (size_t)threadIdx.x * 1280;
  __builtin_prefetch(p, 0, 3);
  __builtin_prefetch(p + 160, 0, 3);
}

// A fragment (16x32, bf16) from per-wave LDS tile, row-major ld=64.
// Layout per ISA 7.12.2: lane m=lane&15, half=lane>>4; VGPR r holds
// K = 8*half + (r<4 ? 2r : 16+2(r-4)) (+0,+1).
__device__ inline v16bf afrag(const __bf16* tile, int k0) {
  int lane = threadIdx.x & 31, m = lane & 15, half = lane >> 4;
  const __bf16* row = tile + m * L_ + k0 + 8 * half;
  v16bf a;
#pragma unroll
  for (int r = 0; r < 8; ++r) {
    int kk = (r < 4) ? 2 * r : 16 + 2 * (r - 4);
    a[2 * r]     = row[kk];
    a[2 * r + 1] = row[kk + 1];
  }
  return a;
}

// B fragment (32x16) from a TRANSPOSED bf16 weight wT[Nout][ldK]:
// lane n=lane&15, half=lane>>4; VGPR r holds K = 8*half + (r<4 ? 2r : 16+2(r-4)) (+0,+1)
// which is contiguous in wT rows -> dword-granular loads.
__device__ inline v16bf bfragT(const __bf16* wT, int ld, int n0, int k0) {
  int lane = threadIdx.x & 31, nn = lane & 15, half = lane >> 4;
  const __bf16* p = wT + (size_t)(n0 + nn) * ld + k0 + half * 8;
  v16bf b;
#pragma unroll
  for (int r = 0; r < 8; ++r) { int kk = (r < 4) ? 2 * r : 16 + 2 * (r - 4); b[2 * r] = p[kk]; b[2 * r + 1] = p[kk + 1]; }
  return b;
}

// store f32 accumulator (16x16 tile, cols col0..col0+15) as bf16 into LDS tile
__device__ inline void store_tile_acc(__bf16* tile, int col0, v8f acc) {
  int lane = threadIdx.x & 31, half = lane >> 4, n = lane & 15;
#pragma unroll
  for (int i = 0; i < 8; ++i)
    tile[(i + 8 * half) * L_ + col0 + n] = (__bf16)acc[i];
}

// stage a 16x64 fp32 tile from global into per-wave LDS as bf16 (float4 loads)
__device__ inline void load_tile_bf16(__bf16* tile, const float* g) {
  int lane = threadIdx.x & 31;
  for (int idx = lane; idx < 16 * 16; idx += 32) {
    int rrow = idx >> 4, c4 = (idx & 15) << 2;
    const float4 f = *reinterpret_cast<const float4*>(g + rrow * L_ + c4);
    __bf16* d2 = tile + rrow * L_ + c4;
    d2[0] = (__bf16)f.x; d2[1] = (__bf16)f.y; d2[2] = (__bf16)f.z; d2[3] = (__bf16)f.w;
  }
}

// ---------------- weight prep: fp32 row-major [K][N] -> bf16 transposed [N][ldK]
__device__ void tconv(const float* W, int K, int Nout, __bf16* wT, int ldT) {
  for (int idx = threadIdx.x; idx < Nout * ldT; idx += blockDim.x) {
    int nn = idx / ldT, kk = idx % ldT;
    wT[idx] = (kk < K) ? (__bf16)W[kk * Nout + nn] : (__bf16)0.0f;
  }
}

__global__ void prep_weights_kernel(
    const float* W_enc, const float* W_gd, const float* gd_Wi, const float* gd_Wh,
    const float* W_dom1, const float* W_dom2, const float* ode_W1, const float* ode_W2,
    const float* W_gc, const float* gc_Wi, const float* gc_Wh, const float* W_dec,
    __bf16* bw) {
  switch (blockIdx.x) {
    case 0:  tconv(W_enc, 16, 64,  bw + OFF_ENC, 32);  break;
    case 1:  tconv(W_gd,  64, 64,  bw + OFF_GD,  64);  break;
    case 2:  tconv(gd_Wi, 64, 192, bw + OFF_GDWI, 64); break;
    case 3:  tconv(gd_Wh, 64, 192, bw + OFF_GDWH, 64); break;
    case 4:  tconv(W_dom1, 64, 64, bw + OFF_DOM1, 64); break;
    case 5:  tconv(W_dom2, 64, 64, bw + OFF_DOM2, 64); break;
    case 6:  tconv(ode_W1, 128, 64, bw + OFF_ODE1, 128); break;
    case 7:  tconv(ode_W2, 64, 64, bw + OFF_ODE2, 64); break;
    case 8:  tconv(W_gc,  64, 64,  bw + OFF_GC,  64);  break;
    case 9:  tconv(gc_Wi, 64, 192, bw + OFF_GCWI, 64); break;
    case 10: tconv(gc_Wh, 64, 192, bw + OFF_GCWH, 64); break;
    case 11: tconv(W_dec, 64, 16,  bw + OFF_DEC, 64);  break;
    default: break;
  }
}

__global__ void zero_kernel(float4* p, int n4) {
  int i = blockIdx.x * blockDim.x + threadIdx.x;
  if (i < n4) { float4 z = {0.f, 0.f, 0.f, 0.f}; p[i] = z; }
}

// ---------------- CSR build (deterministic, atomic-free; runs once) ----------
__global__ void csr_count_kernel(const int* __restrict__ dst, int* __restrict__ counts) {
  int v = blockIdx.x * blockDim.x + threadIdx.x;
  if (v >= V_) return;
  int c = 0;
  for (int e = 0; e < E_; ++e) c += (dst[e] == v) ? 1 : 0;
  counts[v] = c;
}

__global__ void csr_scan_kernel(const int* __restrict__ counts, int* __restrict__ offs) {
  if (blockIdx.x == 0 && threadIdx.x == 0) {
    int acc = 0;
    for (int v = 0; v < V_; ++v) { offs[v] = acc; acc += counts[v]; }
    offs[V_] = acc;
  }
}

__global__ void csr_fill_kernel(const int* __restrict__ src, const int* __restrict__ dst,
                                const float* __restrict__ attr, const int* __restrict__ offs,
                                int* __restrict__ csr_src, float* __restrict__ csr_w) {
  int v = blockIdx.x * blockDim.x + threadIdx.x;
  if (v >= V_) return;
  int p = offs[v];
  for (int e = 0; e < E_; ++e) {
    if (dst[e] == v) { csr_src[p] = src[e]; csr_w[p] = attr[e]; ++p; }
  }
}

// ---------------- atomic-free graph aggregation (replaces zero+scatter):
// one wave per (n, dst-node) row; lane accumulates 2 of 64 cols over in-edges.
__global__ __launch_bounds__(256) void gather_kernel(
    const float* __restrict__ xe, const int* __restrict__ offs,
    const int* __restrict__ csr_src, const float* __restrict__ csr_w,
    float* __restrict__ agg) {
  int gwave = (blockIdx.x * blockDim.x + threadIdx.x) >> 5;
  int lane = threadIdx.x & 31;
  if (gwave >= N_ * V_) return;
  int nb = gwave >> 11;          // V_ == 1<<11
  int v  = gwave & (V_ - 1);
  int beg = offs[v], end = offs[v + 1];
  const size_t base = (size_t)nb * V_ * L_;
  float2 acc = {0.f, 0.f};
  for (int k = beg; k < end; ++k) {
    int s = csr_src[k];
    float w = csr_w[k];
    const float2 xv = *reinterpret_cast<const float2*>(xe + base + (size_t)s * L_ + lane * 2);
    acc.x += xv.x * w;
    acc.y += xv.y * w;
  }
  *reinterpret_cast<float2*>(agg + base + (size_t)v * L_ + lane * 2) = acc;
}

// ---------------- encoder: xe[nv][l] = x[nv, :, t] @ W_enc + b_enc (K=16 padded to 32)
__global__ __launch_bounds__(BLK_) void encode_kernel(
    const float* __restrict__ x, const __bf16* __restrict__ bw,
    const float* __restrict__ b_enc, float* __restrict__ xe, int t) {
  int wave = threadIdx.x >> 5, lane = threadIdx.x & 31;
  int half = lane >> 4, n = lane & 15;
  int node0 = blockIdx.x * NPB_ + wave * 16;
  const __bf16* We = bw + OFF_ENC;
  const float* xp = x + (size_t)(node0 + n) * C_ * T_ + t;  // row m = lane&15
  v16bf a;
#pragma unroll
  for (int r = 0; r < 8; ++r) {
    if (r < 4) {
      int c0 = 8 * half + 2 * r;               // channel index (K), valid < 16
      a[2 * r]     = (__bf16)xp[(size_t)c0 * T_];
      a[2 * r + 1] = (__bf16)xp[(size_t)(c0 + 1) * T_];
    } else { a[2 * r] = (__bf16)0.0f; a[2 * r + 1] = (__bf16)0.0f; }
  }
#pragma unroll
  for (int j = 0; j < 4; ++j) {
    v8f acc = vzero();
    acc = WMMA(a, bfragT(We, 32, j * 16, 0), acc);
    float be = b_enc[j * 16 + n];
#pragma unroll
    for (int i = 0; i < 8; ++i) {
      int node = node0 + i + 8 * half;
      xe[(size_t)node * L_ + j * 16 + n] = acc[i] + be;
    }
  }
}

// ---------------- domain RNN step: m = agg@W_gd; hd = GRU(m, hd)
__global__ __launch_bounds__(BLK_) void domain_gru_kernel(
    const float* __restrict__ agg, float* __restrict__ hd,
    const __bf16* __restrict__ bw, const float* __restrict__ gd_b) {
  __shared__ __bf16 sm[WAVES_][3][16 * L_];
  int wave = threadIdx.x >> 5, lane = threadIdx.x & 31;
  int half = lane >> 4, n = lane & 15;
  int node0 = blockIdx.x * NPB_ + wave * 16;
  prefetch_weights(bw);
  __bf16 *tA = sm[wave][0], *tH = sm[wave][1], *tM = sm[wave][2];
  load_tile_bf16(tA, agg + (size_t)node0 * L_);
  load_tile_bf16(tH, hd + (size_t)node0 * L_);
  const __bf16 *Wgd = bw + OFF_GD, *Wi = bw + OFF_GDWI, *Wh = bw + OFF_GDWH;

  v16bf a0 = afrag(tA, 0), a1 = afrag(tA, 32);
#pragma unroll
  for (int j = 0; j < 4; ++j) {                        // m = agg @ W_gd
    v8f acc = vzero();
    acc = WMMA(a0, bfragT(Wgd, 64, j * 16, 0), acc);
    acc = WMMA(a1, bfragT(Wgd, 64, j * 16, 32), acc);
    store_tile_acc(tM, j * 16, acc);
  }
  v16bf m0 = afrag(tM, 0), m1 = afrag(tM, 32);
  v16bf h0 = afrag(tH, 0), h1 = afrag(tH, 32);
#pragma unroll
  for (int j = 0; j < 4; ++j) {
    int col = j * 16 + n;
    v8f az = vzero(), ar = vzero(), ai = vzero(), ah = vzero();
    az = WMMA(m0, bfragT(Wi, 64, j * 16, 0), az);
    az = WMMA(m1, bfragT(Wi, 64, j * 16, 32), az);
    az = WMMA(h0, bfragT(Wh, 64, j * 16, 0), az);
    az = WMMA(h1, bfragT(Wh, 64, j * 16, 32), az);
    ar = WMMA(m0, bfragT(Wi, 64, 64 + j * 16, 0), ar);
    ar = WMMA(m1, bfragT(Wi, 64, 64 + j * 16, 32), ar);
    ar = WMMA(h0, bfragT(Wh, 64, 64 + j * 16, 0), ar);
    ar = WMMA(h1, bfragT(Wh, 64, 64 + j * 16, 32), ar);
    ai = WMMA(m0, bfragT(Wi, 64, 128 + j * 16, 0), ai);
    ai = WMMA(m1, bfragT(Wi, 64, 128 + j * 16, 32), ai);
    ah = WMMA(h0, bfragT(Wh, 64, 128 + j * 16, 0), ah);
    ah = WMMA(h1, bfragT(Wh, 64, 128 + j * 16, 32), ah);
    float bz = gd_b[col], br = gd_b[64 + col], bn = gd_b[128 + col];
#pragma unroll
    for (int i = 0; i < 8; ++i) {
      int node = node0 + i + 8 * half;
      float z = sigm(az[i] + bz);
      float r = sigm(ar[i] + br);
      float nn2 = tanhf(ai[i] + bn + r * ah[i]);
      float hv = hd[(size_t)node * L_ + col];
      hd[(size_t)node * L_ + col] = (1.0f - z) * nn2 + z * hv;
    }
  }
}

// ---------------- z_D = tanh(hd@W_dom1 + b1) @ W_dom2 + b2
__global__ __launch_bounds__(BLK_) void domain_mlp_kernel(
    const float* __restrict__ hd, float* __restrict__ zD,
    const __bf16* __restrict__ bw, const float* __restrict__ b1,
    const float* __restrict__ b2) {
  __shared__ __bf16 sm[WAVES_][2][16 * L_];
  int wave = threadIdx.x >> 5, lane = threadIdx.x & 31;
  int half = lane >> 4, n = lane & 15;
  int node0 = blockIdx.x * NPB_ + wave * 16;
  __bf16 *tH = sm[wave][0], *tT = sm[wave][1];
  load_tile_bf16(tH, hd + (size_t)node0 * L_);
  const __bf16 *W1 = bw + OFF_DOM1, *W2 = bw + OFF_DOM2;
  v16bf h0 = afrag(tH, 0), h1 = afrag(tH, 32);
#pragma unroll
  for (int j = 0; j < 4; ++j) {
    v8f acc = vzero();
    acc = WMMA(h0, bfragT(W1, 64, j * 16, 0), acc);
    acc = WMMA(h1, bfragT(W1, 64, j * 16, 32), acc);
    float bb = b1[j * 16 + n];
    v8f u;
#pragma unroll
    for (int i = 0; i < 8; ++i) u[i] = tanhf(acc[i] + bb);
    store_tile_acc(tT, j * 16, u);
  }
  v16bf t0 = afrag(tT, 0), t1 = afrag(tT, 32);
#pragma unroll
  for (int j = 0; j < 4; ++j) {
    v8f acc = vzero();
    acc = WMMA(t0, bfragT(W2, 64, j * 16, 0), acc);
    acc = WMMA(t1, bfragT(W2, 64, j * 16, 32), acc);
    float bb = b2[j * 16 + n];
#pragma unroll
    for (int i = 0; i < 8; ++i) {
      int node = node0 + i + 8 * half;
      zD[(size_t)node * L_ + j * 16 + n] = acc[i] + bb;
    }
  }
}

// ---------------- decode a [NV][L] buffer into out[:,:,:,t]
__global__ __launch_bounds__(BLK_) void decode_kernel(
    const float* __restrict__ zbuf, const __bf16* __restrict__ bw,
    const float* __restrict__ b_dec, float* __restrict__ out, int t) {
  __shared__ __bf16 sm[WAVES_][16 * L_];
  int wave = threadIdx.x >> 5, lane = threadIdx.x & 31;
  int half = lane >> 4, n = lane & 15;
  int node0 = blockIdx.x * NPB_ + wave * 16;
  __bf16* tZ = sm[wave];
  load_tile_bf16(tZ, zbuf + (size_t)node0 * L_);
  const __bf16* Wd = bw + OFF_DEC;
  v16bf z0 = afrag(tZ, 0), z1 = afrag(tZ, 32);
  v8f acc = vzero();
  acc = WMMA(z0, bfragT(Wd, 64, 0, 0), acc);
  acc = WMMA(z1, bfragT(Wd, 64, 0, 32), acc);
  float bd = b_dec[n];
#pragma unroll
  for (int i = 0; i < 8; ++i) {
    int node = node0 + i + 8 * half;
    out[(size_t)node * C_ * T_ + (size_t)n * T_ + t] = acc[i] + bd;
  }
}

// ---------------- fused correction step:
//   u = tanh([h, zD] @ ode_W1 + b1); h_ode = h + u @ ode_W2 + b2
//   m = agg @ W_gc; h = GRU(m, h_ode);  out[:,:,:,t] = h @ W_dec + b_dec
__global__ __launch_bounds__(BLK_) void correct_kernel(
    const float* __restrict__ agg, float* __restrict__ h,
    const float* __restrict__ zD, const __bf16* __restrict__ bw,
    const float* __restrict__ ode_b1, const float* __restrict__ ode_b2,
    const float* __restrict__ gc_b, const float* __restrict__ b_dec,
    float* __restrict__ out, int t) {
  __shared__ __bf16 sm[WAVES_][6][16 * L_];
  int wave = threadIdx.x >> 5, lane = threadIdx.x & 31;
  int half = lane >> 4, n = lane & 15;
  int node0 = blockIdx.x * NPB_ + wave * 16;
  prefetch_weights(bw);
  __bf16 *tH = sm[wave][0], *tZ = sm[wave][1], *tA = sm[wave][2];
  __bf16 *tU = sm[wave][3], *tM = sm[wave][4], *tO = sm[wave][5];
  load_tile_bf16(tH, h + (size_t)node0 * L_);
  load_tile_bf16(tZ, zD + (size_t)node0 * L_);
  load_tile_bf16(tA, agg + (size_t)node0 * L_);
  const __bf16 *O1 = bw + OFF_ODE1, *O2 = bw + OFF_ODE2;
  const __bf16 *Wgc = bw + OFF_GC, *Wi = bw + OFF_GCWI, *Wh = bw + OFF_GCWH;
  const __bf16 *Wd = bw + OFF_DEC;

  // u = tanh([h, zD] @ ode_W1 + b1), K=128 in 4 chunks
  v16bf hh0 = afrag(tH, 0), hh1 = afrag(tH, 32);
  v16bf zz0 = afrag(tZ, 0), zz1 = afrag(tZ, 32);
#pragma unroll
  for (int j = 0; j < 4; ++j) {
    v8f acc = vzero();
    acc = WMMA(hh0, bfragT(O1, 128, j * 16, 0), acc);
    acc = WMMA(hh1, bfragT(O1, 128, j * 16, 32), acc);
    acc = WMMA(zz0, bfragT(O1, 128, j * 16, 64), acc);
    acc = WMMA(zz1, bfragT(O1, 128, j * 16, 96), acc);
    float bb = ode_b1[j * 16 + n];
    v8f u;
#pragma unroll
    for (int i = 0; i < 8; ++i) u[i] = tanhf(acc[i] + bb);
    store_tile_acc(tU, j * 16, u);
  }
  // h_ode = h + u @ ode_W2 + b2  (kept fp32 in registers, bf16 in tO)
  v16bf u0 = afrag(tU, 0), u1 = afrag(tU, 32);
  v8f ho[4];
#pragma unroll
  for (int j = 0; j < 4; ++j) {
    v8f acc = vzero();
    acc = WMMA(u0, bfragT(O2, 64, j * 16, 0), acc);
    acc = WMMA(u1, bfragT(O2, 64, j * 16, 32), acc);
    float bb = ode_b2[j * 16 + n];
    int col = j * 16 + n;
#pragma unroll
    for (int i = 0; i < 8; ++i) {
      int node = node0 + i + 8 * half;
      ho[j][i] = h[(size_t)node * L_ + col] + acc[i] + bb;
    }
    store_tile_acc(tO, j * 16, ho[j]);
  }
  // m = agg @ W_gc
  v16bf aa0 = afrag(tA, 0), aa1 = afrag(tA, 32);
#pragma unroll
  for (int j = 0; j < 4; ++j) {
    v8f acc = vzero();
    acc = WMMA(aa0, bfragT(Wgc, 64, j * 16, 0), acc);
    acc = WMMA(aa1, bfragT(Wgc, 64, j * 16, 32), acc);
    store_tile_acc(tM, j * 16, acc);
  }
  // GRU(m, h_ode): h_new written to global h and (bf16) into tU for decode
  v16bf m0 = afrag(tM, 0), m1 = afrag(tM, 32);
  v16bf o0 = afrag(tO, 0), o1 = afrag(tO, 32);
#pragma unroll
  for (int j = 0; j < 4; ++j) {
    int col = j * 16 + n;
    v8f az = vzero(), ar = vzero(), ai = vzero(), ah = vzero();
    az = WMMA(m0, bfragT(Wi, 64, j * 16, 0), az);
    az = WMMA(m1, bfragT(Wi, 64, j * 16, 32), az);
    az = WMMA(o0, bfragT(Wh, 64, j * 16, 0), az);
    az = WMMA(o1, bfragT(Wh, 64, j * 16, 32), az);
    ar = WMMA(m0, bfragT(Wi, 64, 64 + j * 16, 0), ar);
    ar = WMMA(m1, bfragT(Wi, 64, 64 + j * 16, 32), ar);
    ar = WMMA(o0, bfragT(Wh, 64, 64 + j * 16, 0), ar);
    ar = WMMA(o1, bfragT(Wh, 64, 64 + j * 16, 32), ar);
    ai = WMMA(m0, bfragT(Wi, 64, 128 + j * 16, 0), ai);
    ai = WMMA(m1, bfragT(Wi, 64, 128 + j * 16, 32), ai);
    ah = WMMA(o0, bfragT(Wh, 64, 128 + j * 16, 0), ah);
    ah = WMMA(o1, bfragT(Wh, 64, 128 + j * 16, 32), ah);
    float bz = gc_b[col], br = gc_b[64 + col], bn = gc_b[128 + col];
#pragma unroll
    for (int i = 0; i < 8; ++i) {
      int node = node0 + i + 8 * half;
      float z = sigm(az[i] + bz);
      float r = sigm(ar[i] + br);
      float nn2 = tanhf(ai[i] + bn + r * ah[i]);
      float hn = (1.0f - z) * nn2 + z * ho[j][i];
      h[(size_t)node * L_ + col] = hn;
      tU[(i + 8 * half) * L_ + col] = (__bf16)hn;   // reuse tU for decode
    }
  }
  // decode: out[:,:,c,t] = h_new @ W_dec + b_dec  (C=16 -> one column tile)
  v16bf d0 = afrag(tU, 0), d1 = afrag(tU, 32);
  v8f dacc = vzero();
  dacc = WMMA(d0, bfragT(Wd, 64, 0, 0), dacc);
  dacc = WMMA(d1, bfragT(Wd, 64, 0, 32), dacc);
  float bd = b_dec[n];
#pragma unroll
  for (int i = 0; i < 8; ++i) {
    int node = node0 + i + 8 * half;
    out[(size_t)node * C_ * T_ + (size_t)n * T_ + t] = dacc[i] + bd;
  }
}

extern "C" void kernel_launch(void* const* d_in, const int* in_sizes, int n_in,
                              void* d_out, int out_size, void* d_ws, size_t ws_size,
                              hipStream_t stream) {
  const float* x      = (const float*)d_in[0];
  const int*   ei     = (const int*)d_in[1];
  const float* attr   = (const float*)d_in[2];
  const float* W_enc  = (const float*)d_in[3];
  const float* b_enc  = (const float*)d_in[4];
  const float* W_gd   = (const float*)d_in[5];
  const float* gd_Wi  = (const float*)d_in[6];
  const float* gd_Wh  = (const float*)d_in[7];
  const float* gd_b   = (const float*)d_in[8];
  const float* W_dom1 = (const float*)d_in[9];
  const float* b_dom1 = (const float*)d_in[10];
  const float* W_dom2 = (const float*)d_in[11];
  const float* b_dom2 = (const float*)d_in[12];
  const float* ode_W1 = (const float*)d_in[13];
  const float* ode_b1 = (const float*)d_in[14];
  const float* ode_W2 = (const float*)d_in[15];
  const float* ode_b2 = (const float*)d_in[16];
  const float* W_gc   = (const float*)d_in[17];
  const float* gc_Wi  = (const float*)d_in[18];
  const float* gc_Wh  = (const float*)d_in[19];
  const float* gc_b   = (const float*)d_in[20];
  const float* W_dec  = (const float*)d_in[21];
  const float* b_dec  = (const float*)d_in[22];
  float* out = (float*)d_out;

  // workspace layout
  char* base  = (char*)d_ws;
  const size_t SB = (size_t)NV_ * L_ * sizeof(float);   // 4 MB state buffer
  float* hd   = (float*)(base + 0 * SB);
  float* hbuf = (float*)(base + 1 * SB);
  float* zD   = (float*)(base + 2 * SB);
  float* agg  = (float*)(base + 3 * SB);
  float* xe   = (float*)(base + 4 * SB);
  __bf16* bw  = (__bf16*)(base + 5 * SB);
  char* p2    = base + 5 * SB + ((BW_ELEMS * 2 + 255) & ~255);
  int*   offs   = (int*)p2;                 // V_+1
  int*   counts = offs + (V_ + 64);
  int*   csrc   = counts + (V_ + 64);       // E_
  float* cw     = (float*)(csrc + E_);      // E_
  const int* src = ei;
  const int* dst = ei + E_;

  prep_weights_kernel<<<12, 256, 0, stream>>>(W_enc, W_gd, gd_Wi, gd_Wh, W_dom1, W_dom2,
                                              ode_W1, ode_W2, W_gc, gc_Wi, gc_Wh, W_dec, bw);
  // deterministic CSR build (once; reused for all 299 propagation steps)
  csr_count_kernel<<<V_ / 256, 256, 0, stream>>>(dst, counts);
  csr_scan_kernel<<<1, 1, 0, stream>>>(counts, offs);
  csr_fill_kernel<<<V_ / 256, 256, 0, stream>>>(src, dst, attr, offs, csrc, cw);

  const int n4 = NV_ * L_ / 4;
  const dim3 gZero((n4 + 255) / 256), bZero(256);
  const dim3 gNodes(NV_ / NPB_), bNodes(BLK_);
  const dim3 gGath((N_ * V_ * 32) / 256), bGath(256);

  zero_kernel<<<gZero, bZero, 0, stream>>>((float4*)hd, n4);

  // domain RNN over all T timesteps
  for (int t = 0; t < T_; ++t) {
    encode_kernel<<<gNodes, bNodes, 0, stream>>>(x, bw, b_enc, xe, t);
    gather_kernel<<<gGath, bGath, 0, stream>>>(xe, offs, csrc, cw, agg);
    domain_gru_kernel<<<gNodes, bNodes, 0, stream>>>(agg, hd, bw, gd_b);
  }
  domain_mlp_kernel<<<gNodes, bNodes, 0, stream>>>(hd, zD, bw, b_dom1, b_dom2);

  // t = 0: z[0] = xe[0]; decode it; initialize h = xe[0]
  encode_kernel<<<gNodes, bNodes, 0, stream>>>(x, bw, b_enc, xe, 0);
  decode_kernel<<<gNodes, bNodes, 0, stream>>>(xe, bw, b_dec, out, 0);
  hipMemcpyAsync(hbuf, xe, (size_t)NV_ * L_ * sizeof(float),
                 hipMemcpyDeviceToDevice, stream);

  // main ODE + correction loop
  for (int t = 1; t < T_; ++t) {
    encode_kernel<<<gNodes, bNodes, 0, stream>>>(x, bw, b_enc, xe, t);
    gather_kernel<<<gGath, bGath, 0, stream>>>(xe, offs, csrc, cw, agg);
    correct_kernel<<<gNodes, bNodes, 0, stream>>>(agg, hbuf, zD, bw, ode_b1, ode_b2,
                                                  gc_b, b_dec, out, t);
  }
}
